// NEMHSA_24945170055219
// MI455X (gfx1250) — matrix operand order repaired and checked by reading, not compiled, not run on hardware
//
#include <hip/hip_runtime.h>
#include <hip/hip_bf16.h>

#define B_    4
#define T_    2048
#define D_    1024
#define E_    4
#define H_    8
#define DH_   128
#define NTOK_ 512     // T_/E_
#define EPS_  1e-5f

typedef __attribute__((ext_vector_type(16))) _Float16 v16h;
typedef __attribute__((ext_vector_type(8)))  _Float16 v8h;
typedef __attribute__((ext_vector_type(8)))  float    v8f;

// ---------------- WMMA fragment helpers (CDNA5 16x16x32 f16 layouts) -------

// A-matrix 16x32 f16, source row-major (row stride = lda halfs).
__device__ __forceinline__ v16h load_A16(const _Float16* base, int lda) {
    const int l   = threadIdx.x & 31;
    const int row = l & 15;
    const int hi  = l >> 4;
    const _Float16* r = base + (size_t)row * lda;
    union { v16h v; v8h h[2]; } u;
    u.h[0] = *(const v8h*)(r + hi * 8);
    u.h[1] = *(const v8h*)(r + 16 + hi * 8);
    return u.v;
}

// B-matrix 32x16 f16 where element (k,n) lives at base[n*ldn + k] ("n-major").
__device__ __forceinline__ v16h load_Bn(const _Float16* base, int ldn) {
    const int l  = threadIdx.x & 31;
    const int n  = l & 15;
    const int hi = l >> 4;
    const _Float16* r = base + (size_t)n * ldn;
    union { v16h v; v8h h[2]; } u;
    u.h[0] = *(const v8h*)(r + hi * 8);
    u.h[1] = *(const v8h*)(r + 16 + hi * 8);
    return u.v;
}

__device__ __forceinline__ v8f wmma_f16(v16h a, v16h b, v8f c) {
    return __builtin_amdgcn_wmma_f32_16x16x32_f16(false, a, false, b,
                                                  (short)0, c, false, false);
}

__device__ __forceinline__ v8f zero8() {
    v8f z = {0.f,0.f,0.f,0.f,0.f,0.f,0.f,0.f};
    return z;
}

// ---------------- CDNA5 async global->LDS copy (ASYNCcnt path) -------------

__device__ __forceinline__ void async_ld16(unsigned int lds_byte_off, const void* g) {
    asm volatile("global_load_async_to_lds_b128 %0, %1, off"
                 :: "v"(lds_byte_off), "v"((unsigned long long)(uintptr_t)g)
                 : "memory");
}

__device__ __forceinline__ void wait_async0() {
#if __has_builtin(__builtin_amdgcn_s_wait_asynccnt)
    __builtin_amdgcn_s_wait_asynccnt(0);
#else
    asm volatile("s_wait_asynccnt 0x0" ::: "memory");
#endif
}

// ---------------- Kernel 0: cast weights f32 -> f16 ------------------------

__global__ __launch_bounds__(256)
void castw_kernel(const float* __restrict__ qw, const float* __restrict__ kw,
                  const float* __restrict__ vw, const float* __restrict__ ow,
                  _Float16* __restrict__ qw16, _Float16* __restrict__ kw16,
                  _Float16* __restrict__ vw16, _Float16* __restrict__ ow16) {
    int i = blockIdx.x * 256 + threadIdx.x;
    if (i < D_ * D_) {
        qw16[i] = (_Float16)qw[i];
        kw16[i] = (_Float16)kw[i];
        vw16[i] = (_Float16)vw[i];
        ow16[i] = (_Float16)ow[i];
    }
}

// ---------------- Kernel 1: LayerNorm -> f16 -------------------------------

__global__ __launch_bounds__(256)
void ln_kernel(const float* __restrict__ x, const float* __restrict__ nw,
               const float* __restrict__ nb, _Float16* __restrict__ x16) {
    const int row = blockIdx.x;
    const int tid = threadIdx.x;
    const float* xr = x + (size_t)row * D_;
    float v[4];
    float s = 0.f;
    #pragma unroll
    for (int k = 0; k < 4; ++k) { v[k] = xr[tid + k * 256]; s += v[k]; }
    __shared__ float red[8];
    #pragma unroll
    for (int m = 1; m < 32; m <<= 1) s += __shfl_xor(s, m, 32);
    const int wid = tid >> 5, lane = tid & 31;
    if (lane == 0) red[wid] = s;
    __syncthreads();
    float tot = 0.f;
    #pragma unroll
    for (int i = 0; i < 8; ++i) tot += red[i];
    const float mu = tot * (1.0f / D_);
    float s2 = 0.f;
    #pragma unroll
    for (int k = 0; k < 4; ++k) { float d = v[k] - mu; s2 += d * d; }
    #pragma unroll
    for (int m = 1; m < 32; m <<= 1) s2 += __shfl_xor(s2, m, 32);
    __syncthreads();
    if (lane == 0) red[wid] = s2;
    __syncthreads();
    float tv = 0.f;
    #pragma unroll
    for (int i = 0; i < 8; ++i) tv += red[i];
    const float rinv = rsqrtf(tv * (1.0f / D_) + EPS_);
    _Float16* o = x16 + (size_t)row * D_;
    #pragma unroll
    for (int k = 0; k < 4; ++k) {
        int c = tid + k * 256;
        o[c] = (_Float16)((v[k] - mu) * rinv * nw[c] + nb[c]);
    }
}

// ---------------- Kernel 2: QKV GEMMs (ping-pong pipelined) ----------------
// grid: (BT/32, D/256, 3) ; block: 256 = 8 waves (2 M-tiles x 4 N-tiles)

__global__ __launch_bounds__(256)
void qkv_kernel(const _Float16* __restrict__ x16,
                const _Float16* __restrict__ qw, const _Float16* __restrict__ kw,
                const _Float16* __restrict__ vw,
                const float* __restrict__ qb, const float* __restrict__ kb,
                const float* __restrict__ vb,
                _Float16* __restrict__ Q16, _Float16* __restrict__ K16,
                _Float16* __restrict__ Vt16) {
    const int z = blockIdx.z;
    const _Float16* w16  = (z == 0) ? qw : (z == 1) ? kw : vw;
    const float*    bias = (z == 0) ? qb : (z == 1) ? kb : vb;

    const int tid = threadIdx.x, wid = tid >> 5, l = tid & 31;
    const int wm = wid >> 2, wn = wid & 3;
    // wave-uniform values -> force into SGPRs (scalar loop control, SADDR bases)
    const int g0 = __builtin_amdgcn_readfirstlane(blockIdx.x * 32 + wm * 16);
    const int n0 = __builtin_amdgcn_readfirstlane(blockIdx.y * 256 + wn * 64);
    const int b  = g0 / T_;
    const int t0 = g0 % T_;
    const int e  = t0 / NTOK_;
    const int nk = __builtin_amdgcn_readfirstlane((D_ >> e) >> 5);  // 4..32, even

    const _Float16* Ab = x16 + (size_t)g0 * D_;
    const _Float16* Wb = w16 + (size_t)n0 * D_;

    v8f acc[4];
    #pragma unroll
    for (int j = 0; j < 4; ++j) acc[j] = zero8();

    // ping-pong pipeline, trip count even -> no rotation copies
    v16h a0, a1, bf0[4], bf1[4];
    a0 = load_A16(Ab, D_);
    #pragma unroll
    for (int j = 0; j < 4; ++j) bf0[j] = load_Bn(Wb + (size_t)(j * 16) * D_, D_);

    for (int s = 0; s < nk; s += 2) {
        const int k1 = (s + 1) << 5;
        __builtin_prefetch(Ab + k1 + 64, 0, 3);
        a1 = load_A16(Ab + k1, D_);
        #pragma unroll
        for (int j = 0; j < 4; ++j)
            bf1[j] = load_Bn(Wb + (size_t)(j * 16) * D_ + k1, D_);
        #pragma unroll
        for (int j = 0; j < 4; ++j) acc[j] = wmma_f16(a0, bf0[j], acc[j]);
        if (s + 2 < nk) {
            const int k2 = (s + 2) << 5;
            a0 = load_A16(Ab + k2, D_);
            #pragma unroll
            for (int j = 0; j < 4; ++j)
                bf0[j] = load_Bn(Wb + (size_t)(j * 16) * D_ + k2, D_);
        }
        #pragma unroll
        for (int j = 0; j < 4; ++j) acc[j] = wmma_f16(a1, bf1[j], acc[j]);
    }

    const int hi = l >> 4, cn = l & 15;
    #pragma unroll
    for (int j = 0; j < 4; ++j) {
        const int col = n0 + j * 16 + cn;
        const float bv = bias[col];
        const int h = col >> 7, c = col & 127;
        if (z < 2) {
            _Float16* dst = (z == 0) ? Q16 : K16;
            #pragma unroll
            for (int i = 0; i < 8; ++i) {
                int t = t0 + hi * 8 + i;
                dst[((size_t)(b * H_ + h) * T_ + t) * DH_ + c] = (_Float16)(acc[j][i] + bv);
            }
        } else {
            v8h pk;
            #pragma unroll
            for (int i = 0; i < 8; ++i) pk[i] = (_Float16)(acc[j][i] + bv);
            size_t base = ((size_t)(b * H_ + h) * DH_ + c) * T_ + t0 + hi * 8;
            *(v8h*)(Vt16 + base) = pk;   // 8 consecutive t -> one 16B store
        }
    }
}

// ---------------- Kernel 3: flash attention with async double-buffered LDS -
// grid: (B*H, T/128) ; block 256 = 8 waves, each wave owns 16 q-rows.

__global__ __launch_bounds__(256)
void attn_kernel(const _Float16* __restrict__ Q16,
                 const _Float16* __restrict__ K16,
                 const _Float16* __restrict__ Vt16,
                 _Float16* __restrict__ AO16) {
    const int bh = blockIdx.x;
    const int b = bh / H_, h = bh % H_;
    const int tid = threadIdx.x, wid = tid >> 5, l = tid & 31;
    const int hi = l >> 4, ln = l & 15;
    const int q0 = __builtin_amdgcn_readfirstlane(blockIdx.y * 128 + wid * 16);
    const float scale = 0.03125f;                  // D^-0.5 (reference uses D!)

    __shared__ _Float16 Kt[2][32 * DH_];           // (key, c) row-major, 8KB each
    __shared__ _Float16 Vs[2][DH_ * 32];           // (c, key) row-major, 8KB each
    __shared__ _Float16 Pl[8][16 * 32];            // per-wave P staging

    const _Float16* Kg = K16  + (size_t)bh * T_ * DH_;
    const _Float16* Vg = Vt16 + (size_t)bh * DH_ * T_;

    const _Float16* Qb = Q16 + ((size_t)bh * T_ + q0) * DH_;
    v16h qa[4];
    #pragma unroll
    for (int cc = 0; cc < 4; ++cc) qa[cc] = load_A16(Qb + cc * 32, DH_);

    v8f o[8];
    float rm[8], rs[8];
    #pragma unroll
    for (int nt = 0; nt < 8; ++nt) o[nt] = zero8();
    #pragma unroll
    for (int i = 0; i < 8; ++i) { rm[i] = -1e30f; rs[i] = 0.f; }

    _Float16* pw = &Pl[wid][0];

    // stage one 32-key tile (K: 8KB contiguous, V: 128 rows x 64B) into LDS
    auto stage = [&](unsigned kdst, unsigned vdst, int kb) {
        #pragma unroll
        for (int c = 0; c < 2; ++c) {
            const int ch = c * 256 + tid;
            async_ld16(kdst + ch * 16, Kg + (size_t)kb * DH_ + ch * 8);
            const int row = ch >> 2, part = ch & 3;
            async_ld16(vdst + ch * 16, Vg + (size_t)row * T_ + kb + part * 8);
        }
    };

    auto compute = [&](const _Float16* Kl, const _Float16* Vl) {
        v8f s0 = zero8(), s1 = zero8();
        #pragma unroll
        for (int cc = 0; cc < 4; ++cc) {
            v16h b0 = load_Bn(Kl + cc * 32, DH_);
            s0 = wmma_f16(qa[cc], b0, s0);
            v16h b1 = load_Bn(Kl + 16 * DH_ + cc * 32, DH_);
            s1 = wmma_f16(qa[cc], b1, s1);
        }
        #pragma unroll
        for (int i = 0; i < 8; ++i) {
            float a0 = s0[i] * scale, a1 = s1[i] * scale;
            float mx = fmaxf(a0, a1);
            #pragma unroll
            for (int mk = 1; mk < 16; mk <<= 1) mx = fmaxf(mx, __shfl_xor(mx, mk, 16));
            float nm   = fmaxf(rm[i], mx);
            float corr = __expf(rm[i] - nm);
            float p0 = __expf(a0 - nm), p1 = __expf(a1 - nm);
            float ps = p0 + p1;
            #pragma unroll
            for (int mk = 1; mk < 16; mk <<= 1) ps += __shfl_xor(ps, mk, 16);
            rs[i] = rs[i] * corr + ps;
            rm[i] = nm;
            #pragma unroll
            for (int nt = 0; nt < 8; ++nt) o[nt][i] *= corr;
            pw[(hi * 8 + i) * 32 + ln]      = (_Float16)p0;
            pw[(hi * 8 + i) * 32 + 16 + ln] = (_Float16)p1;
        }
        v16h pa = load_A16(pw, 32);       // wave-local LDS RAW: DS in-order
        #pragma unroll
        for (int nt = 0; nt < 8; ++nt) {
            v16h vbf = load_Bn(Vl + (size_t)(nt * 16) * 32, 32);
            o[nt] = wmma_f16(pa, vbf, o[nt]);
        }
    };

    const unsigned k0d = (unsigned)(uintptr_t)&Kt[0][0];
    const unsigned k1d = (unsigned)(uintptr_t)&Kt[1][0];
    const unsigned v0d = (unsigned)(uintptr_t)&Vs[0][0];
    const unsigned v1d = (unsigned)(uintptr_t)&Vs[1][0];

    stage(k0d, v0d, 0);

    const int NIT = T_ / 32;                       // 64, even
    for (int it = 0; it < NIT; it += 2) {
        // ---- buffer 0 ----
        wait_async0();
        __syncthreads();
        stage(k1d, v1d, (it + 1) * 32);            // it+1 < NIT always
        compute(&Kt[0][0], &Vs[0][0]);
        // ---- buffer 1 ----
        wait_async0();
        __syncthreads();
        if (it + 2 < NIT) stage(k0d, v0d, (it + 2) * 32);
        compute(&Kt[1][0], &Vs[1][0]);
    }

    _Float16* Ob = AO16 + ((size_t)b * T_ + q0) * D_ + h * DH_;
    #pragma unroll
    for (int nt = 0; nt < 8; ++nt) {
        #pragma unroll
        for (int i = 0; i < 8; ++i) {
            Ob[(size_t)(hi * 8 + i) * D_ + nt * 16 + ln] = (_Float16)(o[nt][i] / rs[i]);
        }
    }
}

// ---------------- Kernel 4: output projection + bias + residual + pad ------
// grid: (BT/32, D/256) ; block 256 = 8 waves (2 M x 4 N)

__global__ __launch_bounds__(256)
void oproj_kernel(const _Float16* __restrict__ AO16,
                  const _Float16* __restrict__ ow16,
                  const float* __restrict__ ob,
                  const float* __restrict__ x,
                  float* __restrict__ out) {
    const int tid = threadIdx.x, wid = tid >> 5, l = tid & 31;
    const int wm = wid >> 2, wn = wid & 3;
    const int g0 = __builtin_amdgcn_readfirstlane(blockIdx.x * 32 + wm * 16);
    const int n0 = __builtin_amdgcn_readfirstlane(blockIdx.y * 256 + wn * 64);
    const int t0 = g0 % T_;
    const int e  = t0 / NTOK_;
    const int m  = __builtin_amdgcn_readfirstlane(D_ >> e);
    const int nk = m >> 5;                          // 4..32, even

    const _Float16* Ab = AO16 + (size_t)g0 * D_;

    v8f acc[4];
    #pragma unroll
    for (int j = 0; j < 4; ++j) acc[j] = zero8();

    v16h a0, a1, bf0[4], bf1[4];
    a0 = load_A16(Ab, D_);
    #pragma unroll
    for (int j = 0; j < 4; ++j) {
        const int cb = n0 + j * 16;
        if (cb < m) bf0[j] = load_Bn(ow16 + (size_t)cb * D_, D_);
    }

    for (int s = 0; s < nk; s += 2) {
        const int k1 = (s + 1) << 5;
        __builtin_prefetch(Ab + k1 + 64, 0, 3);
        a1 = load_A16(Ab + k1, D_);
        #pragma unroll
        for (int j = 0; j < 4; ++j) {
            const int cb = n0 + j * 16;
            if (cb < m) bf1[j] = load_Bn(ow16 + (size_t)cb * D_ + k1, D_);
        }
        #pragma unroll
        for (int j = 0; j < 4; ++j) {
            const int cb = n0 + j * 16;             // scalar predicate
            if (cb < m) acc[j] = wmma_f16(a0, bf0[j], acc[j]);
        }
        if (s + 2 < nk) {
            const int k2 = (s + 2) << 5;
            a0 = load_A16(Ab + k2, D_);
            #pragma unroll
            for (int j = 0; j < 4; ++j) {
                const int cb = n0 + j * 16;
                if (cb < m) bf0[j] = load_Bn(ow16 + (size_t)cb * D_ + k2, D_);
            }
        }
        #pragma unroll
        for (int j = 0; j < 4; ++j) {
            const int cb = n0 + j * 16;
            if (cb < m) acc[j] = wmma_f16(a1, bf1[j], acc[j]);
        }
    }

    const int hi = l >> 4, cn = l & 15;
    #pragma unroll
    for (int j = 0; j < 4; ++j) {
        const int cb = n0 + j * 16;
        const int col = cb + cn;
        if (cb < m) {
            const float bv = ob[col];
            #pragma unroll
            for (int i = 0; i < 8; ++i) {
                size_t idx = (size_t)(g0 + hi * 8 + i) * D_ + col;
                out[idx] = acc[j][i] + bv + x[idx];
            }
        } else {                                    // zero-padded region: residual only
            #pragma unroll
            for (int i = 0; i < 8; ++i) {
                size_t idx = (size_t)(g0 + hi * 8 + i) * D_ + col;
                out[idx] = x[idx];
            }
        }
    }
}

// ---------------- Host launch ----------------------------------------------

extern "C" void kernel_launch(void* const* d_in, const int* in_sizes, int n_in,
                              void* d_out, int out_size, void* d_ws, size_t ws_size,
                              hipStream_t stream) {
    const float* x  = (const float*)d_in[0];
    // d_in[1] = router_prob (unused by the reference computation)
    const float* nw = (const float*)d_in[2];
    const float* nb = (const float*)d_in[3];
    const float* qw = (const float*)d_in[4];
    const float* qb = (const float*)d_in[5];
    const float* kw = (const float*)d_in[6];
    const float* kb = (const float*)d_in[7];
    const float* vw = (const float*)d_in[8];
    const float* vb = (const float*)d_in[9];
    const float* ow = (const float*)d_in[10];
    const float* ob = (const float*)d_in[11];
    float* out = (float*)d_out;

    const size_t BT = (size_t)B_ * T_;
    _Float16* p = (_Float16*)d_ws;
    _Float16* x16  = p; p += BT * D_;
    _Float16* qw16 = p; p += (size_t)D_ * D_;
    _Float16* kw16 = p; p += (size_t)D_ * D_;
    _Float16* vw16 = p; p += (size_t)D_ * D_;
    _Float16* ow16 = p; p += (size_t)D_ * D_;
    _Float16* Q16  = p; p += BT * D_;
    _Float16* K16  = p; p += BT * D_;
    _Float16* Vt16 = p; p += BT * D_;
    _Float16* AO16 = p;

    castw_kernel<<<(D_ * D_ + 255) / 256, 256, 0, stream>>>(qw, kw, vw, ow,
                                                            qw16, kw16, vw16, ow16);
    ln_kernel<<<(int)BT, 256, 0, stream>>>(x, nw, nb, x16);
    qkv_kernel<<<dim3((int)(BT / 32), D_ / 256, 3), 256, 0, stream>>>(
        x16, qw16, kw16, vw16, qb, kb, vb, Q16, K16, Vt16);
    attn_kernel<<<dim3(B_ * H_, T_ / 128), 256, 0, stream>>>(Q16, K16, Vt16, AO16);
    oproj_kernel<<<dim3((int)(BT / 32), D_ / 256), 256, 0, stream>>>(AO16, ow16, ob, x, out);
}
